// FP8Linear_55456617726139
// MI455X (gfx1250) — matrix operand order repaired and checked by reading, not compiled, not run on hardware
//
#include <hip/hip_runtime.h>
#include <hip/hip_bf16.h>

// out[M,N] = clip( x[M,K] @ W[N,K]^T * (scb[n]/127), +-448 ) -> bf16
// M=1024, N=14336, K=4096.  W int8, x fp32.
//
// Fast path (ws_size >= 126MB): pre-pass converts W->bf16 (117MB, L2-resident)
// and x->bf16 into d_ws, then a conversion-free bf16 WMMA GEMM whose LDS
// staging uses GLOBAL_LOAD_ASYNC_TO_LDS_B128 (ASYNCcnt). Fallback: fused.

typedef __attribute__((ext_vector_type(16))) __bf16 v16bf;
typedef __attribute__((ext_vector_type(8)))  __bf16 v8bf;
typedef __attribute__((ext_vector_type(4)))  __bf16 v4bf;
typedef __attribute__((ext_vector_type(8)))  float  v8f;

#define M_DIM 1024
#define N_DIM 14336
#define K_DIM 4096
#define BM 128
#define BN 256
#define BK 32
#define LDSS 40              // padded LDS K-stride (elements): 80B rows, 16B aligned
#define NKT (K_DIM / BK)     // 128 k-tiles

#if __has_builtin(__builtin_amdgcn_global_load_async_to_lds_b128)
#define ASYNC_STAGE 1
#else
#define ASYNC_STAGE 0
#endif

// exact builtin parameter types: int4 (gcc vector_size) with AS1 / AS3
typedef int i4gcc __attribute__((vector_size(16)));
#define GLB_AS(p) ((__attribute__((address_space(1))) i4gcc*)(p))
#define LDS_AS(p) ((__attribute__((address_space(3))) i4gcc*)(p))

__device__ __forceinline__ void wait_async_zero() {
#if __has_builtin(__builtin_amdgcn_s_wait_asynccnt)
    __builtin_amdgcn_s_wait_asynccnt(0);
#else
    asm volatile("s_wait_asynccnt 0x0" ::: "memory");
#endif
}

// ---------------------------------------------------------------------------
// Pre-pass: W int8 -> bf16   (16 elements / thread)
// ---------------------------------------------------------------------------
__global__ __launch_bounds__(256) void convert_w_bf16(
    const signed char* __restrict__ w, __bf16* __restrict__ wb)
{
    const size_t i = ((size_t)blockIdx.x * 256 + threadIdx.x) * 16;
    union { int4 v; signed char c[16]; } q;
    q.v = *(const int4*)(w + i);
    v8bf a, b;
#pragma unroll
    for (int j = 0; j < 8; ++j) a[j] = (__bf16)(float)q.c[j];
#pragma unroll
    for (int j = 0; j < 8; ++j) b[j] = (__bf16)(float)q.c[8 + j];
    *(v8bf*)(wb + i)     = a;
    *(v8bf*)(wb + i + 8) = b;
}

// Pre-pass: x fp32 -> bf16   (8 elements / thread)
__global__ __launch_bounds__(256) void convert_x_bf16(
    const float* __restrict__ x, __bf16* __restrict__ xb)
{
    const size_t i = ((size_t)blockIdx.x * 256 + threadIdx.x) * 8;
    float4 f0 = *(const float4*)(x + i);
    float4 f1 = *(const float4*)(x + i + 4);
    v8bf v;
    v[0] = (__bf16)f0.x; v[1] = (__bf16)f0.y; v[2] = (__bf16)f0.z; v[3] = (__bf16)f0.w;
    v[4] = (__bf16)f1.x; v[5] = (__bf16)f1.y; v[6] = (__bf16)f1.z; v[7] = (__bf16)f1.w;
    *(v8bf*)(xb + i) = v;
}

// ---------------------------------------------------------------------------
// Main GEMM, conversion-free: A and B already bf16 [row, K] row-major.
// Block 128(M) x 256(N), 8 waves (2x4), each wave 4x4 tiles of 16x16.
// ---------------------------------------------------------------------------
__global__ __launch_bounds__(256) void gemm_bf16_wmma(
    const __bf16* __restrict__ A,     // [M, K] bf16
    const __bf16* __restrict__ B,     // [N, K] bf16
    const float*  __restrict__ scb,
    __bf16* __restrict__ out)
{
    __shared__ __bf16 lA[2][BM * LDSS];   // 2 x 10 KB
    __shared__ __bf16 lB[2][BN * LDSS];   // 2 x 20 KB

    const int tid  = threadIdx.x;
    const int wave = tid >> 5;
    const int lane = tid & 31;
    const int l15  = lane & 15;
    const int hi   = lane >> 4;
    const int wm   = wave & 1;
    const int wn   = wave >> 1;
    const int Mbase = blockIdx.y * BM;
    const int Nbase = blockIdx.x * BN;

    // staging coordinates: 4 threads per 32-elem row, 64 rows per pass
    const int crow = tid >> 2;
    const int ccol = (tid & 3) * 8;

    const __bf16* aptr = A + (size_t)(Mbase + crow) * K_DIM + ccol;
    const __bf16* bptr = B + (size_t)(Nbase + crow) * K_DIM + ccol;

#if ASYNC_STAGE
    // Direct memory -> LDS async copy; no VGPR data, tracked by ASYNCcnt.
    auto issue_tile = [&](int kt, int buf) {
        const int kb = kt * BK;
#pragma unroll
        for (int p = 0; p < 2; ++p)
            __builtin_amdgcn_global_load_async_to_lds_b128(
                GLB_AS(aptr + (size_t)p * 64 * K_DIM + kb),
                LDS_AS(&lA[buf][(p * 64 + crow) * LDSS + ccol]), 0, 0);
#pragma unroll
        for (int p = 0; p < 4; ++p)
            __builtin_amdgcn_global_load_async_to_lds_b128(
                GLB_AS(bptr + (size_t)p * 64 * K_DIM + kb),
                LDS_AS(&lB[buf][(p * 64 + crow) * LDSS + ccol]), 0, 0);
    };
#else
    v8bf apf[2];
    v8bf bpf[4];
    auto load_tile = [&](int kt) {
        const int kb = kt * BK;
#pragma unroll
        for (int p = 0; p < 2; ++p)
            apf[p] = *(const v8bf*)(aptr + (size_t)p * 64 * K_DIM + kb);
#pragma unroll
        for (int p = 0; p < 4; ++p)
            bpf[p] = *(const v8bf*)(bptr + (size_t)p * 64 * K_DIM + kb);
    };
    auto store_tile = [&](int buf) {
#pragma unroll
        for (int p = 0; p < 2; ++p)
            *(v8bf*)&lA[buf][(p * 64 + crow) * LDSS + ccol] = apf[p];
#pragma unroll
        for (int p = 0; p < 4; ++p)
            *(v8bf*)&lB[buf][(p * 64 + crow) * LDSS + ccol] = bpf[p];
    };
#endif

    v8f acc[4][4];
#pragma unroll
    for (int i = 0; i < 4; ++i)
#pragma unroll
        for (int j = 0; j < 4; ++j) acc[i][j] = {};

#if ASYNC_STAGE
    issue_tile(0, 0);
    wait_async_zero();
    __syncthreads();
#else
    load_tile(0);
    store_tile(0);
    __syncthreads();
#endif

    for (int kt = 0; kt < NKT; ++kt) {
        const int cur = kt & 1;
#if ASYNC_STAGE
        if (kt + 1 < NKT) issue_tile(kt + 1, cur ^ 1);
#else
        if (kt + 1 < NKT) load_tile(kt + 1);
#endif

        const __bf16* As = &lA[cur][(wm * 64) * LDSS];
        const __bf16* Bs = &lB[cur][(wn * 64) * LDSS];

        v16bf af[4];
#pragma unroll
        for (int tm = 0; tm < 4; ++tm) {
            const __bf16* p = As + (tm * 16 + l15) * LDSS;
            v8bf lo = *(const v8bf*)(p + hi * 8);
            v8bf hv = *(const v8bf*)(p + 16 + hi * 8);
            af[tm] = __builtin_shufflevector(lo, hv, 0,1,2,3,4,5,6,7,8,9,10,11,12,13,14,15);
        }
#pragma unroll
        for (int tn = 0; tn < 4; ++tn) {
            const __bf16* p = Bs + (tn * 16 + l15) * LDSS;
            v8bf lo = *(const v8bf*)(p + hi * 8);
            v8bf hv = *(const v8bf*)(p + 16 + hi * 8);
            v16bf bfr = __builtin_shufflevector(lo, hv, 0,1,2,3,4,5,6,7,8,9,10,11,12,13,14,15);
#pragma unroll
            for (int tm = 0; tm < 4; ++tm)
                acc[tm][tn] = __builtin_amdgcn_wmma_f32_16x16x32_bf16(
                    false, af[tm], false, bfr, (short)0, acc[tm][tn], false, false);
        }

#if ASYNC_STAGE
        wait_async_zero();
#else
        if (kt + 1 < NKT) store_tile((kt + 1) & 1);
#endif
        __syncthreads();
    }

#pragma unroll
    for (int tn = 0; tn < 4; ++tn) {
        const int n = Nbase + wn * 64 + tn * 16 + l15;
        const float s = scb[n] * (1.0f / 127.0f);
#pragma unroll
        for (int tm = 0; tm < 4; ++tm) {
            const int m0 = Mbase + wm * 64 + tm * 16 + hi * 8;
#pragma unroll
            for (int i = 0; i < 8; ++i) {
                float v = acc[tm][tn][i] * s;
                v = (v != v) ? 448.0f : fminf(fmaxf(v, -448.0f), 448.0f);
                out[(size_t)(m0 + i) * N_DIM + n] = (__bf16)v;
            }
        }
    }
}

// ---------------------------------------------------------------------------
// Fallback: fused conversion-in-staging GEMM (no workspace required)
// ---------------------------------------------------------------------------
__global__ __launch_bounds__(256) void gemm_fused_wmma(
    const float* __restrict__ x,
    const signed char* __restrict__ w,
    const float* __restrict__ scb,
    __bf16* __restrict__ out)
{
    __shared__ __bf16 lA[2][BM * LDSS];
    __shared__ __bf16 lB[2][BN * LDSS];

    const int tid  = threadIdx.x;
    const int wave = tid >> 5;
    const int lane = tid & 31;
    const int l15  = lane & 15;
    const int hi   = lane >> 4;
    const int wm   = wave & 1;
    const int wn   = wave >> 1;
    const int Mbase = blockIdx.y * BM;
    const int Nbase = blockIdx.x * BN;

    const int arow = tid >> 3;
    const int acol = (tid & 7) * 4;
    const int brow = tid >> 2;
    const int bcol = (tid & 3) * 8;

    const float*       aptr = x + (size_t)(Mbase + arow) * K_DIM + acol;
    const signed char* bptr = w + (size_t)(Nbase + brow) * K_DIM + bcol;

    float4 apf[4];
    uint2  bpf[4];

    auto load_tile = [&](int kt) {
        const int kb = kt * BK;
#pragma unroll
        for (int p = 0; p < 4; ++p)
            apf[p] = *(const float4*)(aptr + (size_t)p * 32 * K_DIM + kb);
#pragma unroll
        for (int p = 0; p < 4; ++p)
            bpf[p] = *(const uint2*)(bptr + (size_t)p * 64 * K_DIM + kb);
    };

    auto store_tile = [&](int buf) {
#pragma unroll
        for (int p = 0; p < 4; ++p) {
            v4bf v;
            v.x = (__bf16)apf[p].x;  v.y = (__bf16)apf[p].y;
            v.z = (__bf16)apf[p].z;  v.w = (__bf16)apf[p].w;
            *(v4bf*)&lA[buf][(p * 32 + arow) * LDSS + acol] = v;
        }
#pragma unroll
        for (int p = 0; p < 4; ++p) {
            union { uint2 u; signed char c[8]; } q; q.u = bpf[p];
            v8bf v;
#pragma unroll
            for (int i = 0; i < 8; ++i) v[i] = (__bf16)(float)q.c[i];
            *(v8bf*)&lB[buf][(p * 64 + brow) * LDSS + bcol] = v;
        }
    };

    v8f acc[4][4];
#pragma unroll
    for (int i = 0; i < 4; ++i)
#pragma unroll
        for (int j = 0; j < 4; ++j) acc[i][j] = {};

    load_tile(0);
    store_tile(0);
    __syncthreads();

    for (int kt = 0; kt < NKT; ++kt) {
        const int cur = kt & 1;
        if (kt + 1 < NKT) load_tile(kt + 1);

        const __bf16* As = &lA[cur][(wm * 64) * LDSS];
        const __bf16* Bs = &lB[cur][(wn * 64) * LDSS];

        v16bf af[4];
#pragma unroll
        for (int tm = 0; tm < 4; ++tm) {
            const __bf16* p = As + (tm * 16 + l15) * LDSS;
            v8bf lo = *(const v8bf*)(p + hi * 8);
            v8bf hv = *(const v8bf*)(p + 16 + hi * 8);
            af[tm] = __builtin_shufflevector(lo, hv, 0,1,2,3,4,5,6,7,8,9,10,11,12,13,14,15);
        }
#pragma unroll
        for (int tn = 0; tn < 4; ++tn) {
            const __bf16* p = Bs + (tn * 16 + l15) * LDSS;
            v8bf lo = *(const v8bf*)(p + hi * 8);
            v8bf hv = *(const v8bf*)(p + 16 + hi * 8);
            v16bf bfr = __builtin_shufflevector(lo, hv, 0,1,2,3,4,5,6,7,8,9,10,11,12,13,14,15);
#pragma unroll
            for (int tm = 0; tm < 4; ++tm)
                acc[tm][tn] = __builtin_amdgcn_wmma_f32_16x16x32_bf16(
                    false, af[tm], false, bfr, (short)0, acc[tm][tn], false, false);
        }

        if (kt + 1 < NKT) store_tile((kt + 1) & 1);
        __syncthreads();
    }

#pragma unroll
    for (int tn = 0; tn < 4; ++tn) {
        const int n = Nbase + wn * 64 + tn * 16 + l15;
        const float s = scb[n] * (1.0f / 127.0f);
#pragma unroll
        for (int tm = 0; tm < 4; ++tm) {
            const int m0 = Mbase + wm * 64 + tm * 16 + hi * 8;
#pragma unroll
            for (int i = 0; i < 8; ++i) {
                float v = acc[tm][tn][i] * s;
                v = (v != v) ? 448.0f : fminf(fmaxf(v, -448.0f), 448.0f);
                out[(size_t)(m0 + i) * N_DIM + n] = (__bf16)v;
            }
        }
    }
}

// ---------------------------------------------------------------------------
extern "C" void kernel_launch(void* const* d_in, const int* in_sizes, int n_in,
                              void* d_out, int out_size, void* d_ws, size_t ws_size,
                              hipStream_t stream) {
    const float*       xin = (const float*)d_in[0];
    const signed char* wq  = (const signed char*)d_in[1];
    const float*       scb = (const float*)d_in[2];
    __bf16*            out = (__bf16*)d_out;

    const size_t wElems = (size_t)N_DIM * K_DIM;   // 58,720,256
    const size_t xElems = (size_t)M_DIM * K_DIM;   //  4,194,304
    const size_t need   = (wElems + xElems) * sizeof(__bf16);  // ~126 MB

    dim3 grid(N_DIM / BN, M_DIM / BM);   // (56, 8)
    dim3 block(256);

    if (ws_size >= need) {
        __bf16* wB = (__bf16*)d_ws;          // [N, K] bf16
        __bf16* xB = wB + wElems;            // [M, K] bf16
        convert_w_bf16<<<dim3((unsigned)(wElems / (256 * 16))), block, 0, stream>>>(wq, wB);
        convert_x_bf16<<<dim3((unsigned)(xElems / (256 * 8))),  block, 0, stream>>>(xin, xB);
        gemm_bf16_wmma<<<grid, block, 0, stream>>>(xB, wB, scb, out);
    } else {
        gemm_fused_wmma<<<grid, block, 0, stream>>>(xin, wq, scb, out);
    }
}
